// Attention_79216376808245
// MI455X (gfx1250) — matrix operand order repaired and checked by reading, not compiled, not run on hardware
//
#include <hip/hip_runtime.h>
#include <math.h>

// ---------------------------------------------------------------------------
// GQA attention block for MI455X (gfx1250, wave32, WMMA).
// B=4, S=2048, H=2048, NH=16, NKV=8, HD=128, N_REP=2.
// Matmuls via v_wmma_f32_16x16x32_bf16. Tile staging via gfx1250 async
// LDS-DMA (global_load_async_to_lds_b128 / global_store_async_from_lds_b128,
// ASYNCcnt), double-buffered so the DMA overlaps the WMMA stream.
// ---------------------------------------------------------------------------

typedef __bf16 bf16;
typedef __attribute__((ext_vector_type(16))) __bf16 bf16x16;
typedef __attribute__((ext_vector_type(8)))  __bf16 bf16x8;
typedef __attribute__((ext_vector_type(4)))  __bf16 bf16x4;
typedef __attribute__((ext_vector_type(8)))  float  floatx8;
typedef __attribute__((ext_vector_type(4)))  float  floatx4;
typedef __attribute__((ext_vector_type(4)))  int    intx4;

union BF16x16 { bf16x16 v; bf16x8 h[2]; };

// ---- gfx1250 async LDS-DMA (ASYNCcnt) with compile-safe fallback ----------
#if defined(__has_builtin)
#if __has_builtin(__builtin_amdgcn_global_load_async_to_lds_b128) && \
    __has_builtin(__builtin_amdgcn_global_store_async_from_lds_b128) && \
    __has_builtin(__builtin_amdgcn_s_wait_asynccnt)
#define HAS_ASYNC_LDS 1
#endif
#endif
#ifndef HAS_ASYNC_LDS
#define HAS_ASYNC_LDS 0
#endif

#if HAS_ASYNC_LDS
typedef __attribute__((address_space(1))) intx4 g_intx4;  // global int4
typedef __attribute__((address_space(3))) intx4 l_intx4;  // LDS int4
#endif

__device__ __forceinline__ void cp_g2l_b128(bf16* lds, const bf16* glob) {
#if HAS_ASYNC_LDS
  __builtin_amdgcn_global_load_async_to_lds_b128((g_intx4*)glob, (l_intx4*)lds,
                                                 0, 0);
#else
  *(bf16x8*)lds = *(const bf16x8*)glob;
#endif
}
__device__ __forceinline__ void cp_l2g_b128(void* glob, void* lds) {
#if HAS_ASYNC_LDS
  __builtin_amdgcn_global_store_async_from_lds_b128((g_intx4*)glob,
                                                    (l_intx4*)lds, 0, 0);
#else
  *(intx4*)glob = *(const intx4*)lds;
#endif
}
__device__ __forceinline__ void cp_async_wait() {
#if HAS_ASYNC_LDS
  __builtin_amdgcn_s_wait_asynccnt(0);
#endif
}

__device__ __forceinline__ floatx8 zero8() {
  floatx8 z = {0.f, 0.f, 0.f, 0.f, 0.f, 0.f, 0.f, 0.f};
  return z;
}
__device__ __forceinline__ floatx8 wmma_bf16(bf16x16 a, bf16x16 b, floatx8 c) {
  return __builtin_amdgcn_wmma_f32_16x16x32_bf16(false, a, false, b, (short)0, c,
                                                 false, false);
}

// ------------------------- constants ---------------------------------------
#define BB   4
#define SS   2048
#define HH   2048
#define NHQ  16
#define NKV  8
#define HD   128
#define MTOT (BB * SS)          // 8192
#define NQKV 4096

// ------------------------- fp32 -> bf16 converts (x4 vectorized) -----------
__global__ void cvt_f32_bf16_v4(const float* __restrict__ src, bf16* __restrict__ dst) {
  size_t i = ((size_t)blockIdx.x * blockDim.x + threadIdx.x) * 4;
  floatx4 v = *(const floatx4*)(src + i);
  bf16x4 o = {(bf16)v[0], (bf16)v[1], (bf16)v[2], (bf16)v[3]};
  *(bf16x4*)(dst + i) = o;
}

// Pack Wq|Wk|Wv column-wise into one [2048, 4096] bf16 weight.
__global__ void cvt_wqkv_v4(const float* __restrict__ Wq, const float* __restrict__ Wk,
                            const float* __restrict__ Wv, bf16* __restrict__ dst) {
  size_t i = ((size_t)blockIdx.x * blockDim.x + threadIdx.x) * 4;
  const int col = (int)(i & (NQKV - 1));
  const int row = (int)(i >> 12);
  const float* srcp;
  if (col < 2048)      srcp = Wq + (size_t)row * 2048 + col;
  else if (col < 3072) srcp = Wk + (size_t)row * 1024 + (col - 2048);
  else                 srcp = Wv + (size_t)row * 1024 + (col - 3072);
  floatx4 v = *(const floatx4*)srcp;
  bf16x4 o = {(bf16)v[0], (bf16)v[1], (bf16)v[2], (bf16)v[3]};
  *(bf16x4*)(dst + i) = o;
}

// ------------------------- generic WMMA GEMM --------------------------------
// C[M,N] = A[M,K] bf16 row-major x B[K,N] bf16 row-major.
// Block tile 64x64, 4 waves, K-chunk 64, double-buffered LDS:
//   iter k: issue async A(k+1) + load B(k+1) regs -> compute(k) -> drain B(k+1)
// One barrier per K-chunk; epilogue staged through LDS for b128 stores.
template <bool BF16OUT>
__global__ void __launch_bounds__(128)
gemm_bf16_wmma(const bf16* __restrict__ A, const bf16* __restrict__ B,
               void* __restrict__ Cout, int M, int N, int K) {
  __shared__ __align__(16) char smem[32768];  // 2 x (As 8K + Bt 8K)

  const int tid = threadIdx.x, lane = tid & 31, wave = tid >> 5;
  const int lm = lane & 15, lhi = lane >> 4;
  const int m0 = blockIdx.x * 64, n0 = blockIdx.y * 64;

  floatx8 acc[4];
#pragma unroll
  for (int j = 0; j < 4; ++j) acc[j] = zero8();

  const int arow = tid >> 1;           // 0..63
  const int acol = (tid & 1) * 32;     // 0/32
  const int bkq  = (tid & 15) * 4;     // k quad base 0..60
  const int bnb  = (tid >> 4) * 8;     // n octet base 0..56
  const int mr   = wave * 16 + lm;
  const int koff = lhi * 8;

  const bf16* Abase = A + (size_t)(m0 + arow) * K + acol;
  const bf16* Bbase = B + n0 + bnb;
  const int nk = K >> 6;

  // ---- prologue: stage chunk 0 into buffer 0 ----
  {
    bf16 (*As0)[64] = (bf16(*)[64])smem;
    bf16 (*Bt0)[64] = (bf16(*)[64])(smem + 8192);
    cp_g2l_b128(&As0[arow][acol],      Abase);
    cp_g2l_b128(&As0[arow][acol + 8],  Abase + 8);
    cp_g2l_b128(&As0[arow][acol + 16], Abase + 16);
    cp_g2l_b128(&As0[arow][acol + 24], Abase + 24);
    bf16x8 r[4];
#pragma unroll
    for (int kk = 0; kk < 4; ++kk)
      r[kk] = *(const bf16x8*)(Bbase + (size_t)(bkq + kk) * N);
#pragma unroll
    for (int j = 0; j < 8; ++j) {
      bf16x4 p = {r[0][j], r[1][j], r[2][j], r[3][j]};
      *(bf16x4*)&Bt0[bnb + j][bkq] = p;
    }
  }
  cp_async_wait();
  __syncthreads();

  for (int kc = 0; kc < nk; ++kc) {
    const int cur = kc & 1, nxt = cur ^ 1;
    const int k1 = (kc + 1) << 6;
    const bool pre = (kc + 1 < nk);

    bf16x8 breg[4];
    if (pre) {
      // async A(k+1) into the other buffer (free since last barrier)
      bf16 (*Asn)[64] = (bf16(*)[64])(smem + nxt * 16384);
      const bf16* ag = Abase + k1;
      cp_g2l_b128(&Asn[arow][acol],      ag);
      cp_g2l_b128(&Asn[arow][acol + 8],  ag + 8);
      cp_g2l_b128(&Asn[arow][acol + 16], ag + 16);
      cp_g2l_b128(&Asn[arow][acol + 24], ag + 24);
      // B(k+1) global -> registers (transposed drain after compute)
#pragma unroll
      for (int kk = 0; kk < 4; ++kk)
        breg[kk] = *(const bf16x8*)(Bbase + (size_t)(k1 + bkq + kk) * N);
      if (kc + 2 < nk) {
        __builtin_prefetch(Abase + k1 + 64, 0, 3);
        __builtin_prefetch(Bbase + (size_t)(k1 + 64 + bkq) * N, 0, 3);
      }
    }

    // ---- compute from current buffer: 8 WMMA per wave ----
    {
      bf16 (*Asc)[64] = (bf16(*)[64])(smem + cur * 16384);
      bf16 (*Btc)[64] = (bf16(*)[64])(smem + cur * 16384 + 8192);
#pragma unroll
      for (int c = 0; c < 2; ++c) {
        BF16x16 a;
        a.h[0] = *(const bf16x8*)&Asc[mr][c * 32 + koff];
        a.h[1] = *(const bf16x8*)&Asc[mr][c * 32 + 16 + koff];
#pragma unroll
        for (int j = 0; j < 4; ++j) {
          BF16x16 bfr;
          bfr.h[0] = *(const bf16x8*)&Btc[j * 16 + lm][c * 32 + lhi * 8];
          bfr.h[1] = *(const bf16x8*)&Btc[j * 16 + lm][c * 32 + 16 + lhi * 8];
          acc[j] = wmma_bf16(a.v, bfr.v, acc[j]);
        }
      }
    }

    if (pre) {  // drain B(k+1) registers into the next buffer
      bf16 (*Btn)[64] = (bf16(*)[64])(smem + nxt * 16384 + 8192);
#pragma unroll
      for (int j = 0; j < 8; ++j) {
        bf16x4 p = {breg[0][j], breg[1][j], breg[2][j], breg[3][j]};
        *(bf16x4*)&Btn[bnb + j][bkq] = p;
      }
    }
    cp_async_wait();   // A(k+1) DMA done (latency hidden by the WMMAs)
    __syncthreads();   // everyone's B drain visible; reads of cur complete
  }

  // epilogue: stage fp32 C tile in LDS (aliases buffer 0), coalesced stores
  float* Cs = (float*)smem;  // [64][64]
#pragma unroll
  for (int j = 0; j < 4; ++j)
#pragma unroll
    for (int v = 0; v < 8; ++v)
      Cs[(wave * 16 + v + 8 * lhi) * 64 + j * 16 + lm] = acc[j][v];
  __syncthreads();

  const int rr = tid >> 1, cb = (tid & 1) * 32;
  if (BF16OUT) {
    bf16* Cb = (bf16*)Cout + (size_t)(m0 + rr) * N + n0 + cb;
#pragma unroll
    for (int i = 0; i < 4; ++i) {
      bf16x8 o;
#pragma unroll
      for (int e = 0; e < 8; ++e) o[e] = (bf16)Cs[rr * 64 + cb + 8 * i + e];
      *(bf16x8*)(Cb + 8 * i) = o;
    }
  } else {
    float* Cf = (float*)Cout + (size_t)(m0 + rr) * N + n0 + cb;
#pragma unroll
    for (int i = 0; i < 8; ++i)
      cp_l2g_b128(Cf + 4 * i, &Cs[rr * 64 + cb + 4 * i]);
  }
}

// ------------------------- RoPE + head-major scatter ------------------------
__global__ void rope_scatter(const bf16* __restrict__ qkv,
                             const float* __restrict__ cosT,
                             const float* __restrict__ sinT,
                             bf16* __restrict__ Qh, bf16* __restrict__ Kh,
                             bf16* __restrict__ Vh) {
  size_t i = (size_t)blockIdx.x * blockDim.x + threadIdx.x;  // [0, 8192*4096)
  const int col = (int)(i & (NQKV - 1));
  const int row = (int)(i >> 12);
  const int b = row >> 11;
  const int s = row & (SS - 1);
  const float v = (float)qkv[(size_t)row * NQKV + col];
  if (col < 3072) {
    const int local = (col < 2048) ? col : (col - 2048);
    const int d = local & (HD - 1);
    const int ocol = (d < 64) ? (col + 64) : (col - 64);
    const float other = (float)qkv[(size_t)row * NQKV + ocol];
    const float rot = (d < 64) ? -other : other;
    const float o = v * cosT[s * HD + d] + rot * sinT[s * HD + d];
    if (col < 2048) {
      const int h = col >> 7;
      Qh[(((size_t)(b * NHQ + h)) * SS + s) * HD + d] = (bf16)o;
    } else {
      const int h = (col - 2048) >> 7;
      Kh[(((size_t)(b * NKV + h)) * SS + s) * HD + d] = (bf16)o;
    }
  } else {
    const int h = (col - 3072) >> 7;
    const int d = col & (HD - 1);
    Vh[(((size_t)(b * NKV + h)) * SS + s) * HD + d] = (bf16)v;
  }
}

// ------------------------- causal flash attention ---------------------------
// grid = (S/64, B*NH), block = 128 (4 waves), 16 queries/wave, key tiles of 32.
// Double-buffered K/V staging: async K DMA + register-held V transpose for
// tile t+1 are issued before computing tile t; one barrier per tile.
__global__ void __launch_bounds__(128)
flash_attn_wmma(const bf16* __restrict__ Qh, const bf16* __restrict__ Kh,
                const bf16* __restrict__ Vh, bf16* __restrict__ att) {
  __shared__ __align__(16) char fsm[36864];  // 2 x (Kt 8K + Vt 8K) + Pst 4K
  bf16* Pst = (bf16*)(fsm + 32768);          // [4][16][32]

  const int tid = threadIdx.x, lane = tid & 31, wave = tid >> 5;
  const int lm = lane & 15, lhi = lane >> 4;
  const int bh = blockIdx.y, b = bh >> 4, h = bh & (NHQ - 1), kvh = h >> 1;
  const int q0 = blockIdx.x * 64;
  const float scale = 0.08838834764831845f;  // 1/sqrt(128)

  const bf16* Qbase = Qh + ((size_t)(b * NHQ + h)) * SS * HD;
  const bf16* Kbase = Kh + ((size_t)(b * NKV + kvh)) * SS * HD;
  const bf16* Vbase = Vh + ((size_t)(b * NKV + kvh)) * SS * HD;

  BF16x16 qf[4];
  {
    const int q = q0 + wave * 16 + lm;
#pragma unroll
    for (int c = 0; c < 4; ++c) {
      const bf16* p = Qbase + (size_t)q * HD + c * 32 + lhi * 8;
      qf[c].h[0] = *(const bf16x8*)p;
      qf[c].h[1] = *(const bf16x8*)(p + 16);
    }
  }

  floatx8 o[8];
  float mi[8], li[8];
#pragma unroll
  for (int j = 0; j < 8; ++j) o[j] = zero8();
#pragma unroll
  for (int v = 0; v < 8; ++v) { mi[v] = -1e30f; li[v] = 0.f; }

  const int key = tid >> 2, dblk = (tid & 3) * 32;  // K staging map
  const int vkq = tid & 7, vdb = (tid >> 3) * 8;    // V transpose map
  const int nkt = blockIdx.x * 2 + 2;               // causal: keys [0, q0+64)

  // ---- prologue: stage key tile 0 into buffer 0 ----
  {
    bf16 (*Kt0)[HD] = (bf16(*)[HD])fsm;
    bf16 (*Vt0)[32] = (bf16(*)[32])(fsm + 8192);
    const bf16* kp = Kbase + (size_t)key * HD + dblk;
    cp_g2l_b128(&Kt0[key][dblk],      kp);
    cp_g2l_b128(&Kt0[key][dblk + 8],  kp + 8);
    cp_g2l_b128(&Kt0[key][dblk + 16], kp + 16);
    cp_g2l_b128(&Kt0[key][dblk + 24], kp + 24);
    bf16x8 r[4];
#pragma unroll
    for (int kk = 0; kk < 4; ++kk)
      r[kk] = *(const bf16x8*)(Vbase + (size_t)(4 * vkq + kk) * HD + vdb);
#pragma unroll
    for (int j = 0; j < 8; ++j) {
      bf16x4 p = {r[0][j], r[1][j], r[2][j], r[3][j]};
      *(bf16x4*)&Vt0[vdb + j][4 * vkq] = p;
    }
  }
  cp_async_wait();
  __syncthreads();

  for (int kt = 0; kt < nkt; ++kt) {
    const int cur = kt & 1, nxt = cur ^ 1;
    const int k0 = kt * 32;
    const bool pre = (kt + 1 < nkt);

    bf16x8 vreg[4];
    if (pre) {  // issue K(t+1) async DMA + V(t+1) global->register loads
      bf16 (*Ktn)[HD] = (bf16(*)[HD])(fsm + nxt * 16384);
      const bf16* kp = Kbase + (size_t)(k0 + 32 + key) * HD + dblk;
      cp_g2l_b128(&Ktn[key][dblk],      kp);
      cp_g2l_b128(&Ktn[key][dblk + 8],  kp + 8);
      cp_g2l_b128(&Ktn[key][dblk + 16], kp + 16);
      cp_g2l_b128(&Ktn[key][dblk + 24], kp + 24);
#pragma unroll
      for (int kk = 0; kk < 4; ++kk)
        vreg[kk] =
            *(const bf16x8*)(Vbase + (size_t)(k0 + 32 + 4 * vkq + kk) * HD + vdb);
    }

    bf16 (*Kt)[HD] = (bf16(*)[HD])(fsm + cur * 16384);
    bf16 (*Vt)[32] = (bf16(*)[32])(fsm + cur * 16384 + 8192);

    // S = Q K^T, two 16-key subtiles (8 WMMA)
    floatx8 sacc[2];
#pragma unroll
    for (int sub = 0; sub < 2; ++sub) {
      sacc[sub] = zero8();
#pragma unroll
      for (int c = 0; c < 4; ++c) {
        BF16x16 kb;
        const bf16* kp = &Kt[sub * 16 + lm][c * 32 + lhi * 8];
        kb.h[0] = *(const bf16x8*)kp;
        kb.h[1] = *(const bf16x8*)(kp + 16);
        sacc[sub] = wmma_bf16(qf[c].v, kb.v, sacc[sub]);
      }
    }

    // online softmax across the 16-lane row groups
#pragma unroll
    for (int v = 0; v < 8; ++v) {
      const int qrow = q0 + wave * 16 + v + 8 * lhi;
      float s0 = sacc[0][v] * scale;
      float s1 = sacc[1][v] * scale;
      if (k0 + lm > qrow)      s0 = -1e30f;
      if (k0 + 16 + lm > qrow) s1 = -1e30f;
      float mx = fmaxf(s0, s1);
#pragma unroll
      for (int off = 1; off < 16; off <<= 1)
        mx = fmaxf(mx, __shfl_xor(mx, off, 32));
      const float mnew  = fmaxf(mi[v], mx);
      const float alpha = __expf(mi[v] - mnew);
      const float p0 = __expf(s0 - mnew);
      const float p1 = __expf(s1 - mnew);
      float rs = p0 + p1;
#pragma unroll
      for (int off = 1; off < 16; off <<= 1)
        rs += __shfl_xor(rs, off, 32);
      li[v] = li[v] * alpha + rs;
      mi[v] = mnew;
      Pst[(wave * 16 + v + 8 * lhi) * 32 + lm]      = (bf16)p0;
      Pst[(wave * 16 + v + 8 * lhi) * 32 + 16 + lm] = (bf16)p1;
#pragma unroll
      for (int j = 0; j < 8; ++j) o[j][v] *= alpha;
    }

    // P as A-fragment (16x32), V as B-fragments: 8 WMMA
    BF16x16 pa;
    pa.h[0] = *(const bf16x8*)&Pst[(wave * 16 + lm) * 32 + lhi * 8];
    pa.h[1] = *(const bf16x8*)&Pst[(wave * 16 + lm) * 32 + 16 + lhi * 8];
#pragma unroll
    for (int j = 0; j < 8; ++j) {
      BF16x16 vb;
      const bf16* vp = &Vt[j * 16 + lm][lhi * 8];
      vb.h[0] = *(const bf16x8*)vp;
      vb.h[1] = *(const bf16x8*)(vp + 16);
      o[j] = wmma_bf16(pa.v, vb.v, o[j]);
    }

    if (pre) {  // drain V(t+1) registers into next buffer (4x8 transpose)
      bf16 (*Vtn)[32] = (bf16(*)[32])(fsm + nxt * 16384 + 8192);
#pragma unroll
      for (int j = 0; j < 8; ++j) {
        bf16x4 p = {vreg[0][j], vreg[1][j], vreg[2][j], vreg[3][j]};
        *(bf16x4*)&Vtn[vdb + j][4 * vkq] = p;
      }
    }
    cp_async_wait();   // K(t+1) DMA done; latency hidden by 16 WMMAs
    __syncthreads();
  }

  // epilogue: normalize, stage bf16 tile in LDS (aliases buffer 0), b128 stores
  bf16* Ost = (bf16*)fsm;  // [64][128] 16 KB
#pragma unroll
  for (int v = 0; v < 8; ++v) {
    const float inv = 1.0f / li[v];
#pragma unroll
    for (int j = 0; j < 8; ++j)
      Ost[(wave * 16 + v + 8 * lhi) * 128 + j * 16 + lm] = (bf16)(o[j][v] * inv);
  }
  __syncthreads();
  const int rr = tid >> 1, cb = (tid & 1) * 64;
  bf16* orow = att + ((size_t)(b * SS + q0 + rr)) * HH + h * HD + cb;
#pragma unroll
  for (int i = 0; i < 8; ++i)
    cp_l2g_b128(orow + 8 * i, &Ost[rr * 128 + cb + 8 * i]);
}

// ------------------------- launch ------------------------------------------
extern "C" void kernel_launch(void* const* d_in, const int* in_sizes, int n_in,
                              void* d_out, int out_size, void* d_ws, size_t ws_size,
                              hipStream_t stream) {
  const float* x    = (const float*)d_in[0];
  const float* cosT = (const float*)d_in[1];
  const float* sinT = (const float*)d_in[2];
  const float* Wq   = (const float*)d_in[3];
  const float* Wk   = (const float*)d_in[4];
  const float* Wv   = (const float*)d_in[5];
  const float* Wo   = (const float*)d_in[6];
  float* out = (float*)d_out;

  char* ws = (char*)d_ws;
  bf16* xb   = (bf16*)(ws + 0);              // 33,554,432  (reused as Qh)
  bf16* wqkv = (bf16*)(ws + 33554432);       // 16,777,216
  bf16* wob  = (bf16*)(ws + 50331648);       //  8,388,608
  bf16* qkv  = (bf16*)(ws + 58720256);       // 67,108,864  (reused as att)
  bf16* Kh   = (bf16*)(ws + 125829120);      // 16,777,216
  bf16* Vh   = (bf16*)(ws + 142606336);      // 16,777,216  -> end 159,383,552
  bf16* Qh   = xb;
  bf16* att  = qkv;

  cvt_f32_bf16_v4<<<(MTOT * HH) / 1024, 256, 0, stream>>>(x, xb);
  cvt_wqkv_v4<<<(2048 * NQKV) / 1024, 256, 0, stream>>>(Wq, Wk, Wv, wqkv);
  cvt_f32_bf16_v4<<<(2048 * 2048) / 1024, 256, 0, stream>>>(Wo, wob);

  gemm_bf16_wmma<true><<<dim3(MTOT / 64, NQKV / 64), 128, 0, stream>>>(
      xb, wqkv, qkv, MTOT, NQKV, HH);

  rope_scatter<<<((size_t)MTOT * NQKV) / 256, 256, 0, stream>>>(
      qkv, cosT, sinT, Qh, Kh, Vh);

  flash_attn_wmma<<<dim3(SS / 64, BB * NHQ), 128, 0, stream>>>(Qh, Kh, Vh, att);

  gemm_bf16_wmma<false><<<dim3(MTOT / 64, HH / 64), 128, 0, stream>>>(
      att, wob, out, MTOT, HH, HH);
}
